// DynamicSparseAttention_71004399338083
// MI455X (gfx1250) — matrix operand order repaired and checked by reading, not compiled, not run on hardware
//
#include <hip/hip_runtime.h>

typedef __attribute__((ext_vector_type(16))) _Float16 v16h;
typedef __attribute__((ext_vector_type(8)))  _Float16 v8h;
typedef __attribute__((ext_vector_type(8)))  float    v8f;

#define B_  2
#define S_  2048
#define D_  1024
#define H_  8
#define DH_ 128
#define G2_ 64
#define KTOP_ 204

// ---------------- WMMA helpers (gfx1250 layouts, ISA 7.12.2) ----------------

__device__ __forceinline__ v8f wmma_f16(v16h a, v16h b, v8f c) {
  // emits v_wmma_f32_16x16x32_f16
  return __builtin_amdgcn_wmma_f32_16x16x32_f16(
      /*neg_a=*/false, a, /*neg_b=*/false, b,
      /*c_mod=*/(short)0, c, /*reuse_a=*/false, /*reuse_b=*/false);
}

__device__ __forceinline__ void pf(const void* p) {
  __builtin_prefetch(p, 0, 1);   // emits global_prefetch_b8
}

// A fragment: 16x32 f16, A row-major [M,K], lda in elements.
// lane<16: halves 0..15 = K k0..k0+15
// lane>=16: halves 0..15 = K k0+16..k0+31
__device__ __forceinline__ v16h load_a16(const _Float16* __restrict__ A,
                                         int lda, int m0, int k0) {
  int lane = threadIdx.x & 31;
  int row  = m0 + (lane & 15);
  int kb   = k0 + ((lane >> 4) << 4);
  const _Float16* p = A + (size_t)row * lda + kb;
  v8h lo = *(const v8h*)p;
  v8h hi = *(const v8h*)(p + 8);
  v16h r;
#pragma unroll
  for (int i = 0; i < 8; ++i) { r[i] = lo[i]; r[i + 8] = hi[i]; }
  return r;
}

// B fragment from W^T where W is [N,K] row-major (C = A @ W^T):
// lane gives column n0+(lane&15); 16 contiguous K values (aligned v16h load).
__device__ __forceinline__ v16h load_bt16(const _Float16* __restrict__ W,
                                          int ldw, int n0, int k0) {
  int lane = threadIdx.x & 31;
  int col  = n0 + (lane & 15);
  int kk   = k0 + ((lane >> 4) << 4);
  return *(const v16h*)(W + (size_t)col * ldw + kk);
}

// ---------------- elementwise convert f32 -> f16 ----------------

__global__ void k_cvt(const float* __restrict__ in, _Float16* __restrict__ out, int n) {
  int i = blockIdx.x * blockDim.x + threadIdx.x;
  if (i < n) out[i] = (_Float16)in[i];
}

// ---------------- QKV projection: (x @ W^T + b) * scale -> f16 ----------------
// vtrans==0: out [B,H,S,Dh]   (Q, K)
// vtrans==1: out [B,H,Dh,S]   (V, pre-transposed for the probs@V GEMM)
// grid (M/32, N/512), block 256 (8 waves), wave tile 32x64

__global__ void k_proj(const _Float16* __restrict__ A, const _Float16* __restrict__ W,
                       const float* __restrict__ bias, _Float16* __restrict__ out,
                       float scale, int vtrans) {
  const int K = D_;
  int wave = threadIdx.x >> 5;
  int m0 = blockIdx.x * 32;
  int n0 = blockIdx.y * 512 + wave * 64;
  v8f acc[2][4] = {};
  for (int k0 = 0; k0 < K; k0 += 32) {
    if (k0 + 128 < K) {
      pf(A + (size_t)(m0 + (threadIdx.x & 31)) * K + k0 + 128);
      pf(W + (size_t)(n0 + (threadIdx.x & 15)) * K + k0 + 128);
    }
    v16h a0 = load_a16(A, K, m0, k0);
    v16h a1 = load_a16(A, K, m0 + 16, k0);
#pragma unroll
    for (int f = 0; f < 4; ++f) {
      v16h b = load_bt16(W, K, n0 + f * 16, k0);
      acc[0][f] = wmma_f16(a0, b, acc[0][f]);
      acc[1][f] = wmma_f16(a1, b, acc[1][f]);
    }
  }
  int lane = threadIdx.x & 31;
  int colb = lane & 15;
#pragma unroll
  for (int mi = 0; mi < 2; ++mi) {
    int rowb = m0 + mi * 16 + ((lane >> 4) << 3);
#pragma unroll
    for (int f = 0; f < 4; ++f) {
      int n = n0 + f * 16 + colb;
      int h = n >> 7, d = n & 127;
      float bv = bias[n];
#pragma unroll
      for (int r = 0; r < 8; ++r) {
        int m = rowb + r;
        int b = m >> 11, s = m & (S_ - 1);
        float v = (acc[mi][f][r] + bv) * scale;
        size_t o = vtrans
            ? (((size_t)(b * H_ + h)) * DH_ + d) * S_ + s
            : (((size_t)(b * H_ + h)) * S_ + s) * DH_ + d;
        out[o] = (_Float16)v;
      }
    }
  }
}

// ---------------- scores = Q @ K^T per (b,h) -> f32 [S,S] ----------------
// grid (S/32, S/512), block 256, wave tile 32x64

__global__ void k_scores(const _Float16* __restrict__ Q, const _Float16* __restrict__ Kh,
                         float* __restrict__ scores) {
  int wave = threadIdx.x >> 5;
  int m0 = blockIdx.x * 32;
  int n0 = blockIdx.y * 512 + wave * 64;
  v8f acc[2][4] = {};
  for (int k0 = 0; k0 < DH_; k0 += 32) {
    v16h a0 = load_a16(Q, DH_, m0, k0);
    v16h a1 = load_a16(Q, DH_, m0 + 16, k0);
#pragma unroll
    for (int f = 0; f < 4; ++f) {
      v16h b = load_bt16(Kh, DH_, n0 + f * 16, k0);
      acc[0][f] = wmma_f16(a0, b, acc[0][f]);
      acc[1][f] = wmma_f16(a1, b, acc[1][f]);
    }
  }
  int lane = threadIdx.x & 31;
  int colb = lane & 15;
#pragma unroll
  for (int mi = 0; mi < 2; ++mi) {
    int rowb = m0 + mi * 16 + ((lane >> 4) << 3);
#pragma unroll
    for (int f = 0; f < 4; ++f) {
      int n = n0 + f * 16 + colb;
#pragma unroll
      for (int r = 0; r < 8; ++r)
        scores[(size_t)(rowb + r) * S_ + n] = acc[mi][f][r];
    }
  }
}

// ---------------- top-k threshold (exact radix select) + softmax ------------
// one block (256 threads) per query row

__global__ void k_topk_softmax(const float* __restrict__ scores,
                               _Float16* __restrict__ probs) {
  __shared__ unsigned hist[256];
  __shared__ float red[256];
  __shared__ unsigned s_prefix;
  __shared__ int s_kr;
  int t = threadIdx.x;
  const float* srow = scores + (size_t)blockIdx.x * S_;

  // row max
  float mx = -3.402823466e38f;
  for (int i = t; i < S_; i += 256) mx = fmaxf(mx, srow[i]);
  red[t] = mx; __syncthreads();
  for (int o = 128; o > 0; o >>= 1) {
    if (t < o) red[t] = fmaxf(red[t], red[t + o]);
    __syncthreads();
  }
  float rowmax = red[0];
  if (t == 0) { s_prefix = 0u; s_kr = KTOP_; }
  __syncthreads();

  // 4-pass radix select for k-th largest on monotone keys
  for (int shift = 24; shift >= 0; shift -= 8) {
    hist[t] = 0; __syncthreads();
    unsigned prefix = s_prefix;
    int kr = s_kr;
    unsigned pmask = (shift == 24) ? 0u : (0xFFFFFFFFu << (shift + 8));
    for (int i = t; i < S_; i += 256) {
      unsigned u = __float_as_uint(srow[i]);
      unsigned key = (u & 0x80000000u) ? ~u : (u | 0x80000000u);
      if ((key & pmask) == prefix)
        atomicAdd(&hist[(key >> shift) & 255u], 1u);
    }
    __syncthreads();
    if (t == 0) {
      int accum = 0;
      for (int bin = 255; bin >= 0; --bin) {
        int c = (int)hist[bin];
        if (accum + c >= kr) {
          s_prefix = prefix | ((unsigned)bin << shift);
          s_kr = kr - accum;
          break;
        }
        accum += c;
      }
    }
    __syncthreads();
  }
  unsigned key = s_prefix;
  unsigned tb = (key & 0x80000000u) ? (key & 0x7FFFFFFFu) : ~key;
  float thr = __uint_as_float(tb);

  // masked exp-sum then normalized write (ties >= thr kept, matching reference)
  float sum = 0.f;
  for (int i = t; i < S_; i += 256) {
    float v = srow[i];
    sum += (v >= thr) ? __expf(v - rowmax) : 0.f;
  }
  red[t] = sum; __syncthreads();
  for (int o = 128; o > 0; o >>= 1) {
    if (t < o) red[t] += red[t + o];
    __syncthreads();
  }
  float inv = 1.f / red[0];
  _Float16* prow = probs + (size_t)blockIdx.x * S_;
  for (int i = t; i < S_; i += 256) {
    float v = srow[i];
    float e = (v >= thr) ? __expf(v - rowmax) * inv : 0.f;
    prow[i] = (_Float16)e;
  }
}

// ---------------- attn = probs @ Vt^T per (b,h) -> f32 + f16 [B,S,D] --------
// Vt is [Dh, S] for this (b,h): C[q,d] = sum_j P[q,j] * Vt[d,j]  (W^T form)
// grid (S/128), block 256: wave w -> m0 = bx*128+(w>>1)*32, n0 = (w&1)*64

__global__ void k_attn(const _Float16* __restrict__ P, const _Float16* __restrict__ Vt,
                       float* __restrict__ attn, _Float16* __restrict__ attnh,
                       int bidx, int hidx) {
  int wave = threadIdx.x >> 5;
  int m0 = blockIdx.x * 128 + (wave >> 1) * 32;
  int n0 = (wave & 1) * 64;
  v8f acc[2][4] = {};
  for (int k0 = 0; k0 < S_; k0 += 32) {
    if (k0 + 128 < S_) {
      pf(P + (size_t)(m0 + (threadIdx.x & 31)) * S_ + k0 + 128);
      pf(Vt + (size_t)(n0 + (threadIdx.x & 15)) * S_ + k0 + 128);
    }
    v16h a0 = load_a16(P, S_, m0, k0);
    v16h a1 = load_a16(P, S_, m0 + 16, k0);
#pragma unroll
    for (int f = 0; f < 4; ++f) {
      v16h b = load_bt16(Vt, S_, n0 + f * 16, k0);
      acc[0][f] = wmma_f16(a0, b, acc[0][f]);
      acc[1][f] = wmma_f16(a1, b, acc[1][f]);
    }
  }
  int lane = threadIdx.x & 31;
  int colb = lane & 15;
#pragma unroll
  for (int mi = 0; mi < 2; ++mi) {
    int rowb = m0 + mi * 16 + ((lane >> 4) << 3);
#pragma unroll
    for (int f = 0; f < 4; ++f) {
      int d = n0 + f * 16 + colb;
#pragma unroll
      for (int r = 0; r < 8; ++r) {
        int q = rowb + r;
        size_t o = ((size_t)bidx * S_ + q) * D_ + hidx * DH_ + d;
        float v = acc[mi][f][r];
        attn[o] = v;
        attnh[o] = (_Float16)v;
      }
    }
  }
}

// ---------------- distill: attn.reshape(-1,64) @ Wd^T + bd -----------------
// M = B*S*D/64 = 65536, N = K = 64. grid (M/256), block 256, wave 32 rows

__global__ void k_distill(const _Float16* __restrict__ A, const _Float16* __restrict__ Wd,
                          const float* __restrict__ bd, float* __restrict__ out) {
  int wave = threadIdx.x >> 5;
  int m0 = blockIdx.x * 256 + wave * 32;
  v8f acc[2][4] = {};
  for (int k0 = 0; k0 < G2_; k0 += 32) {
    v16h a0 = load_a16(A, G2_, m0, k0);
    v16h a1 = load_a16(A, G2_, m0 + 16, k0);
#pragma unroll
    for (int f = 0; f < 4; ++f) {
      v16h b = load_bt16(Wd, G2_, f * 16, k0);
      acc[0][f] = wmma_f16(a0, b, acc[0][f]);
      acc[1][f] = wmma_f16(a1, b, acc[1][f]);
    }
  }
  int lane = threadIdx.x & 31;
  int colb = lane & 15;
#pragma unroll
  for (int mi = 0; mi < 2; ++mi) {
    int rowb = m0 + mi * 16 + ((lane >> 4) << 3);
#pragma unroll
    for (int f = 0; f < 4; ++f) {
      int n = f * 16 + colb;
      float bv = bd[n];
#pragma unroll
      for (int r = 0; r < 8; ++r)
        out[(size_t)(rowb + r) * G2_ + n] = acc[mi][f][r] + bv;
    }
  }
}

// ---------------- gate path (tiny) ----------------

__global__ void k_mean(const float* __restrict__ x, float* __restrict__ xmean) {
  int i = blockIdx.x * blockDim.x + threadIdx.x;   // B*D threads
  if (i >= B_ * D_) return;
  int b = i >> 10, d = i & (D_ - 1);
  float s = 0.f;
  for (int j = 0; j < S_; ++j) s += x[((size_t)b * S_ + j) * D_ + d];
  xmean[i] = s * (1.f / (float)S_);
}

__global__ void k_gate1(const float* __restrict__ xmean, const float* __restrict__ Wg,
                        const float* __restrict__ bg, float* __restrict__ g) {
  int t = blockIdx.x * blockDim.x + threadIdx.x;   // B*G2 = 128
  if (t >= B_ * G2_) return;
  int b = t >> 6, j = t & 63;
  float s = bg[j];
  for (int d = 0; d < D_; ++d) s += xmean[b * D_ + d] * Wg[(size_t)j * D_ + d];
  g[t] = 1.f / (1.f + __expf(-s));
}

__global__ void k_gate2(const float* __restrict__ g, const float* __restrict__ Wgp,
                        const float* __restrict__ bgp, float* __restrict__ gate) {
  int i = blockIdx.x * blockDim.x + threadIdx.x;   // B*D
  if (i >= B_ * D_) return;
  int b = i >> 10, n = i & (D_ - 1);
  float s = bgp[n];
  for (int j = 0; j < G2_; ++j) s += g[b * G2_ + j] * Wgp[(size_t)n * G2_ + j];
  gate[i] = s;
}

__global__ void k_mix(const float* __restrict__ gate, const float* __restrict__ dist,
                      const float* __restrict__ attn, _Float16* __restrict__ mixedh) {
  int i = blockIdx.x * blockDim.x + threadIdx.x;   // B*S*D
  if (i >= B_ * S_ * D_) return;
  int n = i & (D_ - 1);
  int b = i >> 21;                                  // S*D = 2^21
  float gv = gate[b * D_ + n];
  mixedh[i] = (_Float16)(gv * dist[i] + (1.f - gv) * attn[i]);
}

// ---------------- out = mixed @ Wo^T + bo -> f32 ----------------
// grid (M/32, N/512), block 256, wave tile 32x64

__global__ void k_out(const _Float16* __restrict__ A, const _Float16* __restrict__ W,
                      const float* __restrict__ bias, float* __restrict__ out) {
  const int K = D_;
  int wave = threadIdx.x >> 5;
  int m0 = blockIdx.x * 32;
  int n0 = blockIdx.y * 512 + wave * 64;
  v8f acc[2][4] = {};
  for (int k0 = 0; k0 < K; k0 += 32) {
    if (k0 + 128 < K) {
      pf(A + (size_t)(m0 + (threadIdx.x & 31)) * K + k0 + 128);
      pf(W + (size_t)(n0 + (threadIdx.x & 15)) * K + k0 + 128);
    }
    v16h a0 = load_a16(A, K, m0, k0);
    v16h a1 = load_a16(A, K, m0 + 16, k0);
#pragma unroll
    for (int f = 0; f < 4; ++f) {
      v16h b = load_bt16(W, K, n0 + f * 16, k0);
      acc[0][f] = wmma_f16(a0, b, acc[0][f]);
      acc[1][f] = wmma_f16(a1, b, acc[1][f]);
    }
  }
  int lane = threadIdx.x & 31;
  int colb = lane & 15;
#pragma unroll
  for (int mi = 0; mi < 2; ++mi) {
    int rowb = m0 + mi * 16 + ((lane >> 4) << 3);
#pragma unroll
    for (int f = 0; f < 4; ++f) {
      int n = n0 + f * 16 + colb;
      float bv = bias[n];
#pragma unroll
      for (int r = 0; r < 8; ++r)
        out[(size_t)(rowb + r) * D_ + n] = acc[mi][f][r] + bv;
    }
  }
}

// ---------------- host orchestration ----------------

extern "C" void kernel_launch(void* const* d_in, const int* in_sizes, int n_in,
                              void* d_out, int out_size, void* d_ws, size_t ws_size,
                              hipStream_t stream) {
  (void)in_sizes; (void)n_in; (void)out_size; (void)ws_size;
  const float* x   = (const float*)d_in[0];
  const float* Wq  = (const float*)d_in[1];  const float* bq  = (const float*)d_in[2];
  const float* Wk  = (const float*)d_in[3];  const float* bk  = (const float*)d_in[4];
  const float* Wv  = (const float*)d_in[5];  const float* bv  = (const float*)d_in[6];
  const float* Wo  = (const float*)d_in[7];  const float* bo  = (const float*)d_in[8];
  const float* Wd  = (const float*)d_in[9];  const float* bd  = (const float*)d_in[10];
  const float* Wg  = (const float*)d_in[11]; const float* bg  = (const float*)d_in[12];
  const float* Wgp = (const float*)d_in[13]; const float* bgp = (const float*)d_in[14];
  float* out = (float*)d_out;

  char* wsb = (char*)d_ws;
  size_t off = 0;
  auto alloc = [&](size_t bytes) {
    size_t o = off; off += (bytes + 255) & ~(size_t)255; return o;
  };
  const size_t NX = (size_t)B_ * S_ * D_;          // 4194304
  _Float16* xh     = (_Float16*)(wsb + alloc(NX * 2));
  _Float16* Wqh    = (_Float16*)(wsb + alloc((size_t)D_ * D_ * 2));
  _Float16* Wkh    = (_Float16*)(wsb + alloc((size_t)D_ * D_ * 2));
  _Float16* Wvh    = (_Float16*)(wsb + alloc((size_t)D_ * D_ * 2));
  _Float16* Woh    = (_Float16*)(wsb + alloc((size_t)D_ * D_ * 2));
  _Float16* Wdh    = (_Float16*)(wsb + alloc((size_t)G2_ * G2_ * 2));
  _Float16* Qh     = (_Float16*)(wsb + alloc(NX * 2));
  _Float16* Kh     = (_Float16*)(wsb + alloc(NX * 2));
  _Float16* Vth    = (_Float16*)(wsb + alloc(NX * 2));   // [B,H,Dh,S]
  float*    scores = (float*)   (wsb + alloc((size_t)S_ * S_ * 4));
  _Float16* probs  = (_Float16*)(wsb + alloc((size_t)S_ * S_ * 2));
  float*    attn   = (float*)   (wsb + alloc(NX * 4));
  _Float16* attnh  = (_Float16*)(wsb + alloc(NX * 2));
  float*    xmean  = (float*)   (wsb + alloc((size_t)B_ * D_ * 4));
  float*    gbuf   = (float*)   (wsb + alloc((size_t)B_ * G2_ * 4));
  float*    gate   = (float*)   (wsb + alloc((size_t)B_ * D_ * 4));
  float*    dist   = scores;              // reuse: scores dead after bh loop
  _Float16* mixedh = probs;               // reuse: probs dead after bh loop

  const float scaleQ = 0.088388347648318447f;  // 1/sqrt(128)

  // converts
  k_cvt<<<(unsigned)((NX + 255) / 256), 256, 0, stream>>>(x, xh, (int)NX);
  k_cvt<<<4096, 256, 0, stream>>>(Wq, Wqh, D_ * D_);
  k_cvt<<<4096, 256, 0, stream>>>(Wk, Wkh, D_ * D_);
  k_cvt<<<4096, 256, 0, stream>>>(Wv, Wvh, D_ * D_);
  k_cvt<<<4096, 256, 0, stream>>>(Wo, Woh, D_ * D_);
  k_cvt<<<16, 256, 0, stream>>>(Wd, Wdh, G2_ * G2_);

  // projections (scale folded into Q; V written transposed [B,H,Dh,S])
  dim3 gp(128, 2);
  k_proj<<<gp, 256, 0, stream>>>(xh, Wqh, bq, Qh, scaleQ, 0);
  k_proj<<<gp, 256, 0, stream>>>(xh, Wkh, bk, Kh, 1.0f, 0);
  k_proj<<<gp, 256, 0, stream>>>(xh, Wvh, bv, Vth, 1.0f, 1);

  // per-(b,h): scores -> topk/softmax -> attn
  for (int bh = 0; bh < B_ * H_; ++bh) {
    const _Float16* Qb = Qh  + (size_t)bh * S_ * DH_;
    const _Float16* Kb = Kh  + (size_t)bh * S_ * DH_;
    const _Float16* Vb = Vth + (size_t)bh * DH_ * S_;
    k_scores<<<dim3(64, 4), 256, 0, stream>>>(Qb, Kb, scores);
    k_topk_softmax<<<S_, 256, 0, stream>>>(scores, probs);
    k_attn<<<16, 256, 0, stream>>>(probs, Vb, attn, attnh, bh >> 3, bh & 7);
  }

  // distill (reuses scores buffer as output)
  k_distill<<<256, 256, 0, stream>>>(attnh, Wdh, bd, dist);

  // gate path
  k_mean<<<(B_ * D_ + 255) / 256, 256, 0, stream>>>(x, xmean);
  k_gate1<<<1, 128, 0, stream>>>(xmean, Wg, bg, gbuf);
  k_gate2<<<(B_ * D_ + 255) / 256, 256, 0, stream>>>(gbuf, Wgp, bgp, gate);

  // mix + output projection
  k_mix<<<(unsigned)((NX + 255) / 256), 256, 0, stream>>>(gate, dist, attn, mixedh);
  k_out<<<dim3(128, 2), 256, 0, stream>>>(mixedh, Woh, bo, out);
}
